// PtrNet_41918880809044
// MI455X (gfx1250) — compile-verified, matrix operand order checked by reading, not run on hardware
//
#include <hip/hip_runtime.h>

#define B_    2048
#define SEQ_  128
#define H_    128
#define G4_   512      // 4*H
#define TB_   16       // batch-tile rows per workgroup (WMMA M)
#define INFV  1e8f
#define CLIPV 10.0f

typedef __attribute__((ext_vector_type(16))) __bf16 v16bf;
typedef __attribute__((ext_vector_type(8)))  float  v8f;

static __device__ __forceinline__ __bf16 f2b(float f) { return (__bf16)f; }
static __device__ __forceinline__ float  b2f(__bf16 b) { return (float)b; }
static __device__ __forceinline__ float  sigf(float x) { return 1.0f / (1.0f + expf(-x)); }

// k-offset inside a bf16 WMMA fragment (A: 16x32 MxK, B: 32xN KxN), per ISA 7.12.2
static __device__ __forceinline__ int fragK(int kc, int j, int lane) {
    return kc * 32 + ((j >> 2) << 4) + ((lane >> 4) << 3) + ((j & 3) << 1);
}

static __device__ __forceinline__ unsigned hash3(unsigned a, unsigned b, unsigned c) {
    a += 0x9e3779b9u * b + 0x85ebca6bu * c;
    a ^= a >> 16; a *= 0x7feb352du;
    a ^= a >> 15; a *= 0x846ca68bu;
    a ^= a >> 16;
    return a;
}

// gfx1250 async global->LDS copy (GV mode, 16B per lane) + wait, per ISA 15.18.3
static __device__ __forceinline__ void async_b128_to_lds(unsigned lds_off, unsigned long long gaddr) {
    asm volatile("global_load_async_to_lds_b128 %0, %1, off"
                 :: "v"(lds_off), "v"(gaddr) : "memory");
}
static __device__ __forceinline__ void wait_async0() {
    asm volatile("s_wait_asynccnt 0x0" ::: "memory");
}

// ---------------------------------------------------------------------------
// 1) Embedding: (B*N,3) @ (3,H)^T -> bf16 (memory-bound, VALU)
// ---------------------------------------------------------------------------
__global__ __launch_bounds__(256) void embed_kernel(
    const float* __restrict__ x, const float* __restrict__ W, __bf16* __restrict__ embedB)
{
    int idx = blockIdx.x * 256 + threadIdx.x;           // over B*N*H
    if (idx >= B_ * SEQ_ * H_) return;
    int h = idx & (H_ - 1);
    int r = idx >> 7;
    const float* xr = x + (size_t)r * 3;
    float v = xr[0] * W[h * 3 + 0] + xr[1] * W[h * 3 + 1] + xr[2] * W[h * 3 + 2];
    embedB[idx] = f2b(v);
}

// ---------------------------------------------------------------------------
// 2) Encoder LSTM: persistent WG per 16 batch rows; weights bf16 in LDS (256KB).
//    x-tile for step t+1 is async-copied to LDS while step t's WMMAs run.
// ---------------------------------------------------------------------------
__global__ __launch_bounds__(128) void enc_kernel(
    const __bf16* __restrict__ embedB,
    const float* __restrict__ Wih, const float* __restrict__ Whh,
    const float* __restrict__ bias,
    __bf16* __restrict__ refB, float* __restrict__ hOut, float* __restrict__ cOut)
{
    extern __shared__ char smem[];
    __bf16* sWih  = (__bf16*)smem;                  // 512*128 bf16
    __bf16* sWhh  = sWih + G4_ * H_;                // 512*128 bf16
    float*  sGate = (float*)(sWhh + G4_ * H_);      // 16*512 f32
    __bf16* sH    = (__bf16*)(sGate + TB_ * G4_);   // 16*128 bf16
    __bf16* sX    = sH + TB_ * H_;                  // 2 x 16*128 bf16 (double buffer)

    const int tid = threadIdx.x, lane = tid & 31, wave = tid >> 5;
    const int b0 = blockIdx.x * TB_;

    // stage x-tile for step t into buffer buf: 256 chunks x 16B, 2 per thread
    auto stage_tile = [&](int t, int buf) {
        #pragma unroll
        for (int i = 0; i < 2; i++) {
            int chunk = tid + i * 128;
            int m  = chunk >> 4;                 // row (16 x 16B chunks per row)
            int co = (chunk & 15) * 16;          // byte offset within row
            unsigned long long g = (unsigned long long)(size_t)
                ((const char*)(embedB + ((size_t)(b0 + m) * SEQ_ + t) * H_) + co);
            unsigned l = (unsigned)(size_t)((const char*)(sX + buf * TB_ * H_) + chunk * 16);
            async_b128_to_lds(l, g);
        }
    };

    stage_tile(0, 0);   // overlap first x-tile with weight staging
    for (int i = tid; i < G4_ * H_; i += 128) { sWih[i] = f2b(Wih[i]); sWhh[i] = f2b(Whh[i]); }
    for (int i = tid; i < TB_ * H_; i += 128) sH[i] = f2b(0.f);
    float creg[TB_];
    #pragma unroll
    for (int m = 0; m < TB_; m++) creg[m] = 0.f;

    const int am = lane & 15;
    for (int t = 0; t < SEQ_; ++t) {
        const int buf = t & 1;
        wait_async0();          // this wave's async copies into buf are done
        __syncthreads();        // all waves' copies visible; prev-step readers done
        if (t + 1 < SEQ_) stage_tile(t + 1, buf ^ 1);   // stream next tile behind WMMAs

        // A fragments: x_t (LDS staged) and h (LDS), K=128 -> 4 chunks
        v16bf afX[4], afH[4];
        const __bf16* xr = sX + buf * TB_ * H_ + am * H_;
        #pragma unroll
        for (int kc = 0; kc < 4; kc++)
            #pragma unroll
            for (int j = 0; j < 8; j++) {
                int k0 = fragK(kc, j, lane);
                afX[kc][2*j] = xr[k0];            afX[kc][2*j+1] = xr[k0+1];
                afH[kc][2*j] = sH[am*H_ + k0];    afH[kc][2*j+1] = sH[am*H_ + k0+1];
            }
        // 8 contiguous gate tiles per wave, processed in independent pairs so the
        // scheduler can hide the WMMA->WMMA RAW hazard with real work.
        for (int p = 0; p < 4; p++) {
            const int ncolA = (wave * 8 + 2 * p) * 16 + (lane & 15);
            const int ncolB = ncolA + 16;
            v8f accA, accB;
            const float bA = bias[ncolA], bB = bias[ncolB];
            #pragma unroll
            for (int j = 0; j < 8; j++) { accA[j] = bA; accB[j] = bB; }
            #pragma unroll
            for (int kc = 0; kc < 4; kc++) {
                v16bf bIA, bHA, bIB, bHB;
                #pragma unroll
                for (int j = 0; j < 8; j++) {
                    int k0 = fragK(kc, j, lane);
                    bIA[2*j] = sWih[ncolA*H_ + k0]; bIA[2*j+1] = sWih[ncolA*H_ + k0+1];
                    bIB[2*j] = sWih[ncolB*H_ + k0]; bIB[2*j+1] = sWih[ncolB*H_ + k0+1];
                    bHA[2*j] = sWhh[ncolA*H_ + k0]; bHA[2*j+1] = sWhh[ncolA*H_ + k0+1];
                    bHB[2*j] = sWhh[ncolB*H_ + k0]; bHB[2*j+1] = sWhh[ncolB*H_ + k0+1];
                }
                accA = __builtin_amdgcn_wmma_f32_16x16x32_bf16(false, afX[kc], false, bIA, (short)0, accA, false, false);
                accB = __builtin_amdgcn_wmma_f32_16x16x32_bf16(false, afX[kc], false, bIB, (short)0, accB, false, false);
                accA = __builtin_amdgcn_wmma_f32_16x16x32_bf16(false, afH[kc], false, bHA, (short)0, accA, false, false);
                accB = __builtin_amdgcn_wmma_f32_16x16x32_bf16(false, afH[kc], false, bHB, (short)0, accB, false, false);
            }
            #pragma unroll
            for (int j = 0; j < 8; j++) {
                int m = j + ((lane >> 4) << 3);
                sGate[m * G4_ + ncolA] = accA[j];
                sGate[m * G4_ + ncolB] = accB[j];
            }
        }
        __syncthreads();
        // elementwise LSTM: thread owns one h column across 16 rows
        {
            const int hid = tid;
            #pragma unroll
            for (int m = 0; m < TB_; m++) {
                float ig = sGate[m*G4_ + hid];
                float fg = sGate[m*G4_ + H_   + hid];
                float gg = sGate[m*G4_ + 2*H_ + hid];
                float og = sGate[m*G4_ + 3*H_ + hid];
                float c  = sigf(fg) * creg[m] + sigf(ig) * tanhf(gg);
                float hv = sigf(og) * tanhf(c);
                creg[m] = c;
                sH[m*H_ + hid] = f2b(hv);
                refB[((size_t)(b0 + m) * SEQ_ + t) * H_ + hid] = f2b(hv);
                if (t == SEQ_ - 1) {
                    hOut[(size_t)(b0 + m) * H_ + hid] = hv;
                    cOut[(size_t)(b0 + m) * H_ + hid] = c;
                }
            }
        }
        __syncthreads();
    }
}

// ---------------------------------------------------------------------------
// 3) u2g = ref@Wref^T+bref, u2p = ref@Wref2^T+bref2 : parallel WMMA GEMM,
//    stored bf16 so both tables (134MB) stay L2-resident for the decoder.
// ---------------------------------------------------------------------------
__global__ __launch_bounds__(128) void ref_proj_kernel(
    const __bf16* __restrict__ refB,
    const float* __restrict__ Wref,  const float* __restrict__ bref,
    const float* __restrict__ Wref2, const float* __restrict__ bref2,
    __bf16* __restrict__ u2g, __bf16* __restrict__ u2p)
{
    const int tid = threadIdx.x, lane = tid & 31, wave = tid >> 5;
    const size_t r0 = ((size_t)blockIdx.x * 4 + wave) * 16;   // row tile over B*N
    const int am = lane & 15;
    const __bf16* ar = refB + (r0 + am) * H_;
    v16bf af[4];
    #pragma unroll
    for (int kc = 0; kc < 4; kc++)
        #pragma unroll
        for (int j = 0; j < 8; j++) {
            int k0 = fragK(kc, j, lane);
            af[kc][2*j] = ar[k0]; af[kc][2*j+1] = ar[k0+1];
        }
    for (int nt = 0; nt < H_ / 16; nt++) {
        const int ncol = nt * 16 + (lane & 15);
        v8f aG, aP;
        const float bg = bref[ncol], bp = bref2[ncol];
        #pragma unroll
        for (int j = 0; j < 8; j++) { aG[j] = bg; aP[j] = bp; }
        #pragma unroll
        for (int kc = 0; kc < 4; kc++) {
            v16bf bG, bP;
            #pragma unroll
            for (int j = 0; j < 8; j++) {
                int k0 = fragK(kc, j, lane);
                bG[2*j] = f2b(Wref[ncol*H_ + k0]);  bG[2*j+1] = f2b(Wref[ncol*H_ + k0+1]);
                bP[2*j] = f2b(Wref2[ncol*H_ + k0]); bP[2*j+1] = f2b(Wref2[ncol*H_ + k0+1]);
            }
            aG = __builtin_amdgcn_wmma_f32_16x16x32_bf16(false, af[kc], false, bG, (short)0, aG, false, false);
            aP = __builtin_amdgcn_wmma_f32_16x16x32_bf16(false, af[kc], false, bP, (short)0, aP, false, false);
        }
        #pragma unroll
        for (int j = 0; j < 8; j++) {
            int m = j + ((lane >> 4) << 3);
            u2g[(r0 + m) * H_ + ncol] = f2b(aG[j]);
            u2p[(r0 + m) * H_ + ncol] = f2b(aP[j]);
        }
    }
}

// ---------------------------------------------------------------------------
// 4) Decoder: persistent WG per 16 batch rows; dec LSTM weights bf16 in LDS.
//    Per step: LSTM gates (WMMA) -> glimpse attention -> pointer -> sample.
// ---------------------------------------------------------------------------
__global__ __launch_bounds__(128) void dec_kernel(
    const __bf16* __restrict__ embedB,
    const __bf16* __restrict__ u2g, const __bf16* __restrict__ u2p,
    const float* __restrict__ Wih, const float* __restrict__ Whh, const float* __restrict__ bias,
    const float* __restrict__ Wq,  const float* __restrict__ bq,
    const float* __restrict__ Wq2, const float* __restrict__ bq2,
    const float* __restrict__ Vec, const float* __restrict__ Vec2,
    const float* __restrict__ din0,
    const float* __restrict__ hIn, const float* __restrict__ cIn,
    float* __restrict__ outPI, float* __restrict__ outLL)
{
    extern __shared__ char smem[];
    __bf16* sWih  = (__bf16*)smem;                    // 128KB
    __bf16* sWhh  = sWih + G4_ * H_;                  // 128KB
    float*  sScr  = (float*)(sWhh + G4_ * H_);        // 32KB: gates, then reused u1/sc
    float*  sMask = sScr + TB_ * G4_;                 // 16*128 f32
    __bf16* sH    = (__bf16*)(sMask + TB_ * SEQ_);    // 16*128
    __bf16* sDin  = sH + TB_ * H_;                    // 16*128
    __bf16* sQbf  = sDin + TB_ * H_;                  // 16*128
    float*  sVec  = (float*)(sQbf + TB_ * H_);        // 128
    float*  sVec2 = sVec + H_;                        // 128
    float*  sRmax = sVec2 + H_;                       // 16
    float*  sRsum = sRmax + TB_;                      // 16
    float*  sLL   = sRsum + TB_;                      // 16
    int*    sNxt  = (int*)(sLL + TB_);                // 16

    float* sGate = sScr;             // phase 1 (16x512)
    float* sU1   = sScr;             // phase 2+ (16x128) - gates dead by then
    float* sSc   = sScr + TB_ * H_;  // (16x128)

    const int tid = threadIdx.x, lane = tid & 31, wave = tid >> 5;
    const int b0 = blockIdx.x * TB_;

    for (int i = tid; i < G4_ * H_; i += 128) { sWih[i] = f2b(Wih[i]); sWhh[i] = f2b(Whh[i]); }
    for (int i = tid; i < TB_ * SEQ_; i += 128) sMask[i] = 0.f;
    if (tid < H_) { sVec[tid] = Vec[tid]; sVec2[tid] = Vec2[tid]; }
    if (tid < TB_) sLL[tid] = 0.f;
    float creg[TB_];
    #pragma unroll
    for (int m = 0; m < TB_; m++) {
        creg[m] = cIn[(size_t)(b0 + m) * H_ + tid];
        sH[m*H_ + tid]   = f2b(hIn[(size_t)(b0 + m) * H_ + tid]);
        sDin[m*H_ + tid] = f2b(din0[tid]);
    }
    __syncthreads();

    const int am = lane & 15;
    for (int t = 0; t < SEQ_; ++t) {
        // ---- LSTM gates: dec_in@Wih^T + h@Whh^T + b ----
        v16bf afD[4], afH[4];
        #pragma unroll
        for (int kc = 0; kc < 4; kc++)
            #pragma unroll
            for (int j = 0; j < 8; j++) {
                int k0 = fragK(kc, j, lane);
                afD[kc][2*j] = sDin[am*H_ + k0]; afD[kc][2*j+1] = sDin[am*H_ + k0+1];
                afH[kc][2*j] = sH[am*H_ + k0];   afH[kc][2*j+1] = sH[am*H_ + k0+1];
            }
        for (int p = 0; p < 4; p++) {
            const int ncolA = (wave * 8 + 2 * p) * 16 + (lane & 15);
            const int ncolB = ncolA + 16;
            v8f accA, accB;
            const float bA = bias[ncolA], bB = bias[ncolB];
            #pragma unroll
            for (int j = 0; j < 8; j++) { accA[j] = bA; accB[j] = bB; }
            #pragma unroll
            for (int kc = 0; kc < 4; kc++) {
                v16bf bIA, bHA, bIB, bHB;
                #pragma unroll
                for (int j = 0; j < 8; j++) {
                    int k0 = fragK(kc, j, lane);
                    bIA[2*j] = sWih[ncolA*H_ + k0]; bIA[2*j+1] = sWih[ncolA*H_ + k0+1];
                    bIB[2*j] = sWih[ncolB*H_ + k0]; bIB[2*j+1] = sWih[ncolB*H_ + k0+1];
                    bHA[2*j] = sWhh[ncolA*H_ + k0]; bHA[2*j+1] = sWhh[ncolA*H_ + k0+1];
                    bHB[2*j] = sWhh[ncolB*H_ + k0]; bHB[2*j+1] = sWhh[ncolB*H_ + k0+1];
                }
                accA = __builtin_amdgcn_wmma_f32_16x16x32_bf16(false, afD[kc], false, bIA, (short)0, accA, false, false);
                accB = __builtin_amdgcn_wmma_f32_16x16x32_bf16(false, afD[kc], false, bIB, (short)0, accB, false, false);
                accA = __builtin_amdgcn_wmma_f32_16x16x32_bf16(false, afH[kc], false, bHA, (short)0, accA, false, false);
                accB = __builtin_amdgcn_wmma_f32_16x16x32_bf16(false, afH[kc], false, bHB, (short)0, accB, false, false);
            }
            #pragma unroll
            for (int j = 0; j < 8; j++) {
                int m = j + ((lane >> 4) << 3);
                sGate[m*G4_ + ncolA] = accA[j];
                sGate[m*G4_ + ncolB] = accB[j];
            }
        }
        __syncthreads();
        // ---- LSTM state update ----
        {
            const int hid = tid;
            #pragma unroll
            for (int m = 0; m < TB_; m++) {
                float ig = sGate[m*G4_ + hid];
                float fg = sGate[m*G4_ + H_   + hid];
                float gg = sGate[m*G4_ + 2*H_ + hid];
                float og = sGate[m*G4_ + 3*H_ + hid];
                float c  = sigf(fg) * creg[m] + sigf(ig) * tanhf(gg);
                creg[m] = c;
                sH[m*H_ + hid] = f2b(sigf(og) * tanhf(c));
            }
        }
        __syncthreads();
        // ---- u1 = h @ Wq^T + bq (Wq stays L2-hot; 2 tiles/wave, dual acc) ----
        {
            v16bf ah[4];
            #pragma unroll
            for (int kc = 0; kc < 4; kc++)
                #pragma unroll
                for (int j = 0; j < 8; j++) {
                    int k0 = fragK(kc, j, lane);
                    ah[kc][2*j] = sH[am*H_ + k0]; ah[kc][2*j+1] = sH[am*H_ + k0+1];
                }
            const int ncolA = (wave * 2) * 16 + (lane & 15);
            const int ncolB = ncolA + 16;
            v8f accA, accB;
            const float bA = bq[ncolA], bB = bq[ncolB];
            #pragma unroll
            for (int j = 0; j < 8; j++) { accA[j] = bA; accB[j] = bB; }
            #pragma unroll
            for (int kc = 0; kc < 4; kc++) {
                v16bf bWA, bWB;
                #pragma unroll
                for (int j = 0; j < 8; j++) {
                    int k0 = fragK(kc, j, lane);
                    bWA[2*j] = f2b(Wq[ncolA*H_ + k0]); bWA[2*j+1] = f2b(Wq[ncolA*H_ + k0+1]);
                    bWB[2*j] = f2b(Wq[ncolB*H_ + k0]); bWB[2*j+1] = f2b(Wq[ncolB*H_ + k0+1]);
                }
                accA = __builtin_amdgcn_wmma_f32_16x16x32_bf16(false, ah[kc], false, bWA, (short)0, accA, false, false);
                accB = __builtin_amdgcn_wmma_f32_16x16x32_bf16(false, ah[kc], false, bWB, (short)0, accB, false, false);
            }
            #pragma unroll
            for (int j = 0; j < 8; j++) {
                int m = j + ((lane >> 4) << 3);
                sU1[m*H_ + ncolA] = accA[j];
                sU1[m*H_ + ncolB] = accB[j];
            }
        }
        __syncthreads();
        // ---- glimpse scores: tanh(u1 + u2g) . Vec  (u2g L2-resident bf16) ----
        for (int idx = tid; idx < TB_ * SEQ_; idx += 128) {
            int m = idx >> 7, n = idx & (SEQ_ - 1);
            const __bf16* ur = u2g + (((size_t)(b0 + m)) * SEQ_ + n) * H_;
            if (idx + 128 < TB_ * SEQ_) {
                int m2 = (idx + 128) >> 7, n2 = (idx + 128) & (SEQ_ - 1);
                __builtin_prefetch(u2g + (((size_t)(b0 + m2)) * SEQ_ + n2) * H_, 0, 0);
            }
            float s = 0.f;
            for (int h = 0; h < H_; ++h) s += tanhf(sU1[m*H_ + h] + b2f(ur[h])) * sVec[h];
            sSc[idx] = s - INFV * sMask[idx];
        }
        __syncthreads();
        if (tid < TB_) { float mx = -3.4e38f; for (int n = 0; n < SEQ_; n++) mx = fmaxf(mx, sSc[tid*SEQ_ + n]); sRmax[tid] = mx; }
        __syncthreads();
        for (int idx = tid; idx < TB_ * SEQ_; idx += 128) { int m = idx >> 7; sSc[idx] = expf(sSc[idx] - sRmax[m]); }
        __syncthreads();
        if (tid < TB_) { float s = 0.f; for (int n = 0; n < SEQ_; n++) s += sSc[tid*SEQ_ + n]; sRsum[tid] = 1.f / s; }
        __syncthreads();
        // ---- q = softmax(sc) @ u2g  (per-row GEMV, u2g from L2) ----
        for (int idx = tid; idx < TB_ * H_; idx += 128) {
            int m = idx >> 7, h = idx & (H_ - 1);
            const __bf16* base = u2g + ((size_t)(b0 + m)) * SEQ_ * H_ + h;
            float acc = 0.f;
            for (int n = 0; n < SEQ_; ++n) acc += sSc[m*SEQ_ + n] * b2f(base[(size_t)n * H_]);
            sQbf[idx] = f2b(acc * sRsum[m]);
        }
        __syncthreads();
        // ---- u1p = q @ Wq2^T + bq2 (2 tiles/wave, dual acc) ----
        {
            v16bf aq[4];
            #pragma unroll
            for (int kc = 0; kc < 4; kc++)
                #pragma unroll
                for (int j = 0; j < 8; j++) {
                    int k0 = fragK(kc, j, lane);
                    aq[kc][2*j] = sQbf[am*H_ + k0]; aq[kc][2*j+1] = sQbf[am*H_ + k0+1];
                }
            const int ncolA = (wave * 2) * 16 + (lane & 15);
            const int ncolB = ncolA + 16;
            v8f accA, accB;
            const float bA = bq2[ncolA], bB = bq2[ncolB];
            #pragma unroll
            for (int j = 0; j < 8; j++) { accA[j] = bA; accB[j] = bB; }
            #pragma unroll
            for (int kc = 0; kc < 4; kc++) {
                v16bf bWA, bWB;
                #pragma unroll
                for (int j = 0; j < 8; j++) {
                    int k0 = fragK(kc, j, lane);
                    bWA[2*j] = f2b(Wq2[ncolA*H_ + k0]); bWA[2*j+1] = f2b(Wq2[ncolA*H_ + k0+1]);
                    bWB[2*j] = f2b(Wq2[ncolB*H_ + k0]); bWB[2*j+1] = f2b(Wq2[ncolB*H_ + k0+1]);
                }
                accA = __builtin_amdgcn_wmma_f32_16x16x32_bf16(false, aq[kc], false, bWA, (short)0, accA, false, false);
                accB = __builtin_amdgcn_wmma_f32_16x16x32_bf16(false, aq[kc], false, bWB, (short)0, accB, false, false);
            }
            #pragma unroll
            for (int j = 0; j < 8; j++) {
                int m = j + ((lane >> 4) << 3);
                sU1[m*H_ + ncolA] = accA[j];
                sU1[m*H_ + ncolB] = accB[j];
            }
        }
        __syncthreads();
        // ---- pointer logits: CLIP * tanh(u1p + u2p) . Vec2 - INF*mask ----
        for (int idx = tid; idx < TB_ * SEQ_; idx += 128) {
            int m = idx >> 7, n = idx & (SEQ_ - 1);
            const __bf16* ur = u2p + (((size_t)(b0 + m)) * SEQ_ + n) * H_;
            if (idx + 128 < TB_ * SEQ_) {
                int m2 = (idx + 128) >> 7, n2 = (idx + 128) & (SEQ_ - 1);
                __builtin_prefetch(u2p + (((size_t)(b0 + m2)) * SEQ_ + n2) * H_, 0, 0);
            }
            float s = 0.f;
            for (int h = 0; h < H_; ++h) s += tanhf(sU1[m*H_ + h] + b2f(ur[h])) * sVec2[h];
            sSc[idx] = CLIPV * s - INFV * sMask[idx];
        }
        __syncthreads();
        // ---- sample (deterministic Gumbel-max), log-likelihood, mask, PI ----
        if (tid < TB_) {
            const int m = tid;
            float mx = -3.4e38f;
            for (int n = 0; n < SEQ_; n++) mx = fmaxf(mx, sSc[m*SEQ_ + n]);
            float ssum = 0.f;
            for (int n = 0; n < SEQ_; n++) ssum += expf(sSc[m*SEQ_ + n] - mx);
            const float lse = mx + logf(ssum);
            float best = -3.4e38f; int bi = 0;
            for (int n = 0; n < SEQ_; n++) {
                unsigned hsh = hash3((unsigned)t + 0x1234u, (unsigned)(b0 + m), (unsigned)n);
                float u = (float)(hsh >> 8) * (1.0f / 16777216.0f) + 1e-7f;
                float g = -logf(-logf(u));
                float v = sSc[m*SEQ_ + n] + g;
                if (v > best) { best = v; bi = n; }
            }
            sNxt[m] = bi;
            sLL[m] += sSc[m*SEQ_ + bi] - lse;
            sMask[m*SEQ_ + bi] = 1.f;
            outPI[(size_t)(b0 + m) * SEQ_ + t] = (float)bi;
        }
        __syncthreads();
        // ---- dec_in = embed[b, nxt, :] ----
        for (int idx = tid; idx < TB_ * H_; idx += 128) {
            int m = idx >> 7, h = idx & (H_ - 1);
            sDin[idx] = embedB[((size_t)(b0 + m) * SEQ_ + sNxt[m]) * H_ + h];
        }
        __syncthreads();
    }
    if (tid < TB_) outLL[b0 + tid] = sLL[tid];
}

// ---------------------------------------------------------------------------
extern "C" void kernel_launch(void* const* d_in, const int* in_sizes, int n_in,
                              void* d_out, int out_size, void* d_ws, size_t ws_size,
                              hipStream_t stream) {
    (void)in_sizes; (void)n_in; (void)out_size; (void)ws_size;
    const float* x        = (const float*)d_in[0];
    const float* W_embed  = (const float*)d_in[1];
    const float* enc_Wih  = (const float*)d_in[2];
    const float* enc_Whh  = (const float*)d_in[3];
    const float* enc_b    = (const float*)d_in[4];
    const float* dec_Wih  = (const float*)d_in[5];
    const float* dec_Whh  = (const float*)d_in[6];
    const float* dec_b    = (const float*)d_in[7];
    const float* Wq       = (const float*)d_in[8];
    const float* bq       = (const float*)d_in[9];
    const float* Wq2      = (const float*)d_in[10];
    const float* bq2      = (const float*)d_in[11];
    const float* Wref     = (const float*)d_in[12];
    const float* bref     = (const float*)d_in[13];
    const float* Wref2    = (const float*)d_in[14];
    const float* bref2    = (const float*)d_in[15];
    const float* Vec      = (const float*)d_in[16];
    const float* Vec2     = (const float*)d_in[17];
    const float* din0     = (const float*)d_in[18];

    char* ws = (char*)d_ws;
    size_t off = 0;
    auto alloc = [&](size_t bytes) -> void* {
        void* p = ws + off;
        off = (off + bytes + 255) & ~(size_t)255;
        return p;
    };
    const size_t BNH = (size_t)B_ * SEQ_ * H_;
    __bf16* embedB = (__bf16*)alloc(BNH * 2);
    __bf16* refB   = (__bf16*)alloc(BNH * 2);
    __bf16* u2g    = (__bf16*)alloc(BNH * 2);
    __bf16* u2p    = (__bf16*)alloc(BNH * 2);
    float*  hEnc   = (float*)alloc((size_t)B_ * H_ * 4);
    float*  cEnc   = (float*)alloc((size_t)B_ * H_ * 4);

    float* outPI = (float*)d_out;                 // (B, N)
    float* outLL = outPI + (size_t)B_ * SEQ_;     // (B,)

    // 1) embedding
    embed_kernel<<<(int)((BNH + 255) / 256), 256, 0, stream>>>(x, W_embed, embedB);

    // 2) encoder LSTM (persistent, 16 rows/WG, weights in LDS, async x staging)
    const size_t encLds = (size_t)2 * G4_ * H_ * 2      // weights
                        + (size_t)TB_ * G4_ * 4         // gates
                        + (size_t)TB_ * H_ * 2          // sH
                        + (size_t)2 * TB_ * H_ * 2;     // sX double buffer
    enc_kernel<<<B_ / TB_, 128, encLds, stream>>>(embedB, enc_Wih, enc_Whh, enc_b, refB, hEnc, cEnc);

    // 3) attention reference projections (parallel WMMA GEMM)
    ref_proj_kernel<<<(B_ * SEQ_) / 64, 128, 0, stream>>>(refB, Wref, bref, Wref2, bref2, u2g, u2p);

    // 4) decoder (persistent, 16 rows/WG)
    const size_t decLds = (size_t)2 * G4_ * H_ * 2      // weights
                        + (size_t)TB_ * G4_ * 4         // gates / u1 / sc scratch
                        + (size_t)TB_ * SEQ_ * 4        // mask
                        + (size_t)TB_ * H_ * 2 * 3      // sH, sDin, sQbf
                        + (size_t)H_ * 4 * 2            // Vec, Vec2
                        + (size_t)TB_ * 4 * 4;          // rmax, rsum, ll, nxt
    dec_kernel<<<B_ / TB_, 128, decLds, stream>>>(embedB, u2g, u2p,
                                                  dec_Wih, dec_Whh, dec_b,
                                                  Wq, bq, Wq2, bq2, Vec, Vec2,
                                                  din0, hEnc, cEnc, outPI, outLL);
}
